// MPNNModel_20169166422334
// MI455X (gfx1250) — compile-verified, matrix operand order checked by reading, not run on hardware
//
#include <hip/hip_runtime.h>
#include <hip/hip_bf16.h>
#include <cstdint>

typedef __attribute__((ext_vector_type(16))) __bf16 v16bf;
typedef __attribute__((ext_vector_type(8)))  float  v8f;

union Frag16 { v16bf v; unsigned int u[8]; };

__device__ __forceinline__ __bf16 f2bf(float f) { return (__bf16)f; }

// ---------------- GCN branch ----------------
__global__ void lin_kernel(const float* __restrict__ x, const float* __restrict__ w,
                           float* __restrict__ xw) {
  __shared__ float xs[33];
  int n = blockIdx.x, tid = threadIdx.x;
  if (tid < 33) xs[tid] = x[(size_t)n * 33 + tid];
  __syncthreads();
  const float* wr = w + tid * 33;
  float s = 0.f;
  #pragma unroll
  for (int k = 0; k < 33; ++k) s += wr[k] * xs[k];
  xw[(size_t)n * 256 + tid] = s;
}

__global__ void deg_kernel(const long long* __restrict__ col, float* __restrict__ deg, int E) {
  int e = blockIdx.x * blockDim.x + threadIdx.x;
  if (e < E) atomicAdd(&deg[(int)col[e]], 1.0f);
}

__global__ void msg_kernel(const long long* __restrict__ row, const long long* __restrict__ col,
                           const float* __restrict__ deg, const float* __restrict__ xw,
                           float* __restrict__ agg) {
  int e = blockIdx.x, tid = threadIdx.x;
  int r = (int)row[e], c = (int)col[e];
  float dr = deg[r], dc = deg[c];
  float norm = (dr > 0.f ? rsqrtf(dr) : 0.f) * (dc > 0.f ? rsqrtf(dc) : 0.f);
  atomicAdd(&agg[(size_t)c * 256 + tid], norm * xw[(size_t)r * 256 + tid]);
}

__global__ void drugsum_kernel(const float* __restrict__ agg, const float* __restrict__ lin_b,
                               const long long* __restrict__ batch,
                               float* __restrict__ sums, float* __restrict__ cnt) {
  int n = blockIdx.x, tid = threadIdx.x;
  int b = (int)batch[n];
  atomicAdd(&sums[b * 256 + tid], agg[(size_t)n * 256 + tid] + lin_b[tid]);
  if (tid == 0) atomicAdd(&cnt[b], 1.0f);
}

__global__ void drugfin_kernel(const float* __restrict__ sums, const float* __restrict__ cnt,
                               float* __restrict__ z) {
  int i = blockIdx.x * blockDim.x + threadIdx.x;   // i = b*256+c
  int b = i >> 8, c = i & 255;
  float cn = cnt[b];
  z[b * 1024 + c] = cn > 0.f ? sums[i] / cn : 0.f;
}

// ---------------- CNN branch ----------------
// fp32 [Cout][Cin][KW] -> bf16 [Cout][KW][Cin] (t-major K so k = t*Cin + c,
// with Cin a power of two -> gather math is shift/and only)
__global__ void cvt_w_kernel(const float* __restrict__ in, __bf16* __restrict__ out,
                             int Cin, int KW, int total) {
  int i = blockIdx.x * blockDim.x + threadIdx.x;
  if (i >= total) return;
  int ck = Cin * KW;
  int n = i / ck, r = i - n * ck;
  int c = r / KW, t = r - c * KW;
  out[(size_t)n * ck + t * Cin + c] = f2bf(in[i]);
}

// conv1: 26ch -> 64ch, K=3, direct (tiny), ReLU, bf16 out [B,64,998]
__global__ void conv1_kernel(const float* __restrict__ x, const float* __restrict__ w,
                             const float* __restrict__ bias, __bf16* __restrict__ out,
                             int total) {
  int i = blockIdx.x * blockDim.x + threadIdx.x;
  if (i >= total) return;
  int l = i % 998; int t2 = i / 998;
  int oc = t2 % 64; int b = t2 / 64;
  const float* xb = x + (size_t)b * 26 * 1000 + l;
  const float* wr = w + oc * 78;
  float s = bias[oc];
  #pragma unroll
  for (int c = 0; c < 26; ++c) {
    #pragma unroll
    for (int t = 0; t < 3; ++t) s += xb[c * 1000 + t] * wr[c * 3 + t];
  }
  s = s > 0.f ? s : 0.f;
  out[i] = f2bf(s);
}

// Implicit-im2col bf16 WMMA conv.
// block = 128 threads (4 waves). Each wave owns a 16(L) x 32(Cout) output tile
// (two adjacent N-tiles n0 and n0+16). K-step = 64 (two k-halves), so one LDS
// A-tile fill (16x64 bf16) feeds 4 WMMAs per wave and is shared by 4 waves
// (8 N-tiles). Weights are pre-reordered so k = t*Cin + c with Cin = 2^cshift.
// Epilogue: bias+ReLU, optional bf16 store, optional fused max-pool via
// atomicMax on the (non-negative) float bit pattern.
__global__ void conv_wmma(const __bf16* __restrict__ in, const __bf16* __restrict__ w,
                          const float* __restrict__ bias,
                          __bf16* __restrict__ out, unsigned int* __restrict__ pool,
                          int cshift, int Lin, int Cout, int Lout, int KW) {
  __shared__ __bf16 ldsA[16 * 64];
  const int tid  = threadIdx.x;
  const int lane = tid & 31;
  const int wave = tid >> 5;
  const int lt   = blockIdx.x;                       // L-tile
  const int b    = blockIdx.z;
  const int Cin  = 1 << cshift;
  const int Ktot = Cin * KW;
  const int steps = Ktot >> 6;                       // 64 K per step
  const int nt0  = blockIdx.y * 8 + wave * 2;        // two adjacent N-tiles
  const int n0   = nt0 * 16 + (lane & 15);
  const __bf16* inb = in + (size_t)b * Cin * Lin;
  const unsigned int* w0 = (const unsigned int*)(w + (size_t)n0 * Ktot);
  const unsigned int* w1 = (const unsigned int*)(w + (size_t)(n0 + 16) * Ktot);
  const int m  = lane & 15;
  const int kb = (lane >> 4) * 8;
  const int cmask = Cin - 1;

  v8f acc0 = {};
  v8f acc1 = {};
  for (int s = 0; s < steps; ++s) {
    const int k0 = s * 64;
    __syncthreads();
    #pragma unroll
    for (int i = 0; i < 8; ++i) {                    // 128 thr * 8 = 1024 elems
      int idx = tid * 8 + i;
      int mm = idx >> 6, kk = idx & 63;
      int l  = lt * 16 + mm;
      int kg = k0 + kk;
      int c  = kg & cmask, t = kg >> cshift;         // power-of-two Cin
      __bf16 v = (__bf16)0.0f;
      if (l < Lout) v = inb[(size_t)c * Lin + l + t];
      ldsA[mm * 64 + kk] = v;
    }
    // prefetch next step's weight rows into L2/L0
    if (s + 1 < steps) {
      __builtin_prefetch((const char*)w0 + (size_t)(k0 + 64) * 2, 0, 1);
      __builtin_prefetch((const char*)w1 + (size_t)(k0 + 64) * 2, 0, 1);
    }
    __syncthreads();
    const unsigned int* lds32 = (const unsigned int*)ldsA;
    #pragma unroll
    for (int h = 0; h < 2; ++h) {                    // two 32-K halves
      Frag16 a, b0, b1;
      #pragma unroll
      for (int j = 0; j < 8; ++j) {
        int kj = h * 32 + kb + ((j >> 2) << 4) + ((j & 3) << 1);
        a.u[j]  = lds32[(m * 64 + kj) >> 1];
        b0.u[j] = w0[(k0 + kj) >> 1];
        b1.u[j] = w1[(k0 + kj) >> 1];
      }
      acc0 = __builtin_amdgcn_wmma_f32_16x16x32_bf16(false, a.v, false, b0.v,
                                                     (short)0, acc0, false, false);
      acc1 = __builtin_amdgcn_wmma_f32_16x16x32_bf16(false, a.v, false, b1.v,
                                                     (short)0, acc1, false, false);
    }
  }
  // epilogue
  float bn0 = bias[n0];
  float bn1 = bias[n0 + 16];
  int l0 = lt * 16 + ((lane >> 4) << 3);             // C-layout M rows
  float mx0 = 0.f, mx1 = 0.f;
  __bf16* op0 = out ? out + ((size_t)b * Cout + n0) * Lout : nullptr;
  __bf16* op1 = out ? out + ((size_t)b * Cout + n0 + 16) * Lout : nullptr;
  #pragma unroll
  for (int r = 0; r < 8; ++r) {
    int l = l0 + r;
    float v0 = acc0[r] + bn0; v0 = v0 > 0.f ? v0 : 0.f;
    float v1 = acc1[r] + bn1; v1 = v1 > 0.f ? v1 : 0.f;
    if (l < Lout) {
      if (op0) { op0[l] = f2bf(v0); op1[l] = f2bf(v1); }
      if (v0 > mx0) mx0 = v0;
      if (v1 > mx1) mx1 = v1;
    }
  }
  if (pool) {
    atomicMax(&pool[b * Cout + n0],      __float_as_uint(mx0));
    atomicMax(&pool[b * Cout + n0 + 16], __float_as_uint(mx1));
  }
}

// ---------------- dense head ----------------
__global__ void dense_kernel(const float* __restrict__ in, const float* __restrict__ w,
                             const float* __restrict__ bias, float* __restrict__ out,
                             int K, int O, int ostride, int relu, int total) {
  int i = blockIdx.x * blockDim.x + threadIdx.x;
  if (i >= total) return;
  int b = i / O, o = i - b * O;
  const float* ir = in + (size_t)b * K;
  const float* wr = w + (size_t)o * K;
  float s = bias[o];
  for (int k = 0; k < K; ++k) s += ir[k] * wr[k];
  if (relu) s = s > 0.f ? s : 0.f;
  out[(size_t)b * ostride + o] = s;
}

extern "C" void kernel_launch(void* const* d_in, const int* in_sizes, int n_in,
                              void* d_out, int out_size, void* d_ws, size_t ws_size,
                              hipStream_t stream) {
  (void)n_in; (void)out_size; (void)ws_size;
  const float*     x      = (const float*)d_in[0];
  const long long* ei     = (const long long*)d_in[1];
  const long long* batch  = (const long long*)d_in[2];
  const float*     target = (const float*)d_in[3];
  const float* lin_w = (const float*)d_in[4];
  const float* lin_b = (const float*)d_in[5];
  const float* c1w = (const float*)d_in[6];  const float* c1b = (const float*)d_in[7];
  const float* c2w = (const float*)d_in[8];  const float* c2b = (const float*)d_in[9];
  const float* c3w = (const float*)d_in[10]; const float* c3b = (const float*)d_in[11];
  const float* fcw = (const float*)d_in[12]; const float* fcb = (const float*)d_in[13];
  const float* m1w = (const float*)d_in[14]; const float* m1b = (const float*)d_in[15];
  const float* m2w = (const float*)d_in[16]; const float* m2b = (const float*)d_in[17];
  const float* m3w = (const float*)d_in[18]; const float* m3b = (const float*)d_in[19];
  const float* m4w = (const float*)d_in[20]; const float* m4b = (const float*)d_in[21];

  const int N = in_sizes[0] / 33;            // 128000
  const int E = in_sizes[1] / 2;             // 512000
  const int B = in_sizes[3] / (26 * 1000);   // 128

  char* ws = (char*)d_ws;
  size_t off = 0;
  auto alloc = [&](size_t bytes) -> void* {
    void* p = ws + off;
    off += (bytes + 255) & ~(size_t)255;
    return p;
  };
  float* xw   = (float*)alloc((size_t)N * 256 * 4);
  float* agg  = (float*)alloc((size_t)N * 256 * 4);
  float* deg  = (float*)alloc((size_t)N * 4);
  float* dsum = (float*)alloc((size_t)B * 256 * 4);
  float* cnt  = (float*)alloc((size_t)B * 4);
  __bf16* h1  = (__bf16*)alloc((size_t)B * 64 * 998 * 2);
  __bf16* w2b = (__bf16*)alloc((size_t)256 * 64 * 3 * 2);
  __bf16* h2  = (__bf16*)alloc((size_t)B * 256 * 996 * 2);
  __bf16* w3b = (__bf16*)alloc((size_t)1024 * 256 * 5 * 2);
  unsigned int* pool = (unsigned int*)alloc((size_t)B * 1024 * 4);
  float* z  = (float*)alloc((size_t)B * 1024 * 4);
  float* z1 = (float*)alloc((size_t)B * 1024 * 4);
  float* z2 = (float*)alloc((size_t)B * 1024 * 4);
  float* z3 = (float*)alloc((size_t)B * 512 * 4);

  hipMemsetAsync(deg,  0, (size_t)N * 4, stream);
  hipMemsetAsync(agg,  0, (size_t)N * 256 * 4, stream);
  hipMemsetAsync(dsum, 0, (size_t)B * 256 * 4, stream);
  hipMemsetAsync(cnt,  0, (size_t)B * 4, stream);
  hipMemsetAsync(pool, 0, (size_t)B * 1024 * 4, stream);

  // ---- GCN branch ----
  lin_kernel<<<N, 256, 0, stream>>>(x, lin_w, xw);
  deg_kernel<<<(E + 255) / 256, 256, 0, stream>>>(ei + E, deg, E);
  msg_kernel<<<E, 256, 0, stream>>>(ei, ei + E, deg, xw, agg);
  drugsum_kernel<<<N, 256, 0, stream>>>(agg, lin_b, batch, dsum, cnt);
  drugfin_kernel<<<(B * 256) / 256, 256, 0, stream>>>(dsum, cnt, z);

  // ---- CNN branch ----
  cvt_w_kernel<<<(256 * 64 * 3 + 255) / 256, 256, 0, stream>>>(c2w, w2b, 64, 3, 256 * 64 * 3);
  cvt_w_kernel<<<(1024 * 256 * 5 + 255) / 256, 256, 0, stream>>>(c3w, w3b, 256, 5, 1024 * 256 * 5);
  int t1 = B * 64 * 998;
  conv1_kernel<<<(t1 + 255) / 256, 256, 0, stream>>>(target, c1w, c1b, h1, t1);
  // conv2: Cin=64 (cshift=6), Lin=998 -> Cout=256, Lout=996, KW=3 (Ktot=192, 3 K-steps)
  conv_wmma<<<dim3(63, 2, B), 128, 0, stream>>>(h1, w2b, c2b, h2, nullptr,
                                                6, 998, 256, 996, 3);
  // conv3: Cin=256 (cshift=8), Lin=996 -> Cout=1024, Lout=992, KW=5 (Ktot=1280, 20 K-steps)
  conv_wmma<<<dim3(62, 8, B), 128, 0, stream>>>(h2, w3b, c3b, nullptr, pool,
                                                8, 996, 1024, 992, 5);
  // fc: pool[B,1024] @ fcw^T -> z[:, 256:1024], ReLU
  dense_kernel<<<(B * 768 + 255) / 256, 256, 0, stream>>>(
      (const float*)pool, fcw, fcb, z + 256, 1024, 768, 1024, 1, B * 768);

  // ---- head MLP ----
  dense_kernel<<<(B * 1024 + 255) / 256, 256, 0, stream>>>(z,  m1w, m1b, z1, 1024, 1024, 1024, 1, B * 1024);
  dense_kernel<<<(B * 1024 + 255) / 256, 256, 0, stream>>>(z1, m2w, m2b, z2, 1024, 1024, 1024, 1, B * 1024);
  dense_kernel<<<(B * 512  + 255) / 256, 256, 0, stream>>>(z2, m3w, m3b, z3, 1024, 512,  512,  1, B * 512);
  dense_kernel<<<(B + 255) / 256, 256, 0, stream>>>(z3, m4w, m4b, (float*)d_out, 512, 1, 1, 0, B);
}
